// Conv_14267881357874
// MI455X (gfx1250) — matrix operand order repaired and checked by reading, not compiled, not run on hardware
//
#include <hip/hip_runtime.h>
#include <hip/hip_bf16.h>

typedef __attribute__((ext_vector_type(16))) _Float16 v16h;
typedef __attribute__((ext_vector_type(8)))  float    v8f;

#define N 8192
#define D 128

// ---------------------------------------------------------------------------
// Kernel 1: h = x*conv_w + conv_b ; Wh1 = h@a[:D] ; Wh2 = h@a[D:]
// Store hT[d][i] as f16 (transposed) so WMMA B tiles load contiguously.
// Block = 16 rows x 128 cols, 256 threads.
// ---------------------------------------------------------------------------
__global__ __launch_bounds__(256) void prep_kernel(
    const float* __restrict__ x, const float* __restrict__ conv_w,
    const float* __restrict__ conv_b, const float* __restrict__ a,
    _Float16* __restrict__ hT, float* __restrict__ Wh1, float* __restrict__ Wh2)
{
    __shared__ float hs[16][D];
    __shared__ float s1[16], s2[16];

    const int tid = threadIdx.x;
    const int d   = tid & 127;
    const int rh  = tid >> 7;                 // 0..1 (row half)
    const int rowbase = blockIdx.x * 16;
    const float w = conv_w[0], b = conv_b[0];
    const float a1 = a[d], a2 = a[D + d];

    if (tid < 16) { s1[tid] = 0.f; s2[tid] = 0.f; }
    __syncthreads();

    #pragma unroll
    for (int rr = 0; rr < 8; ++rr) {
        const int r = rh * 8 + rr;
        float hv = x[(size_t)(rowbase + r) * D + d] * w + b;
        hs[r][d] = hv;
        float p1 = hv * a1, p2 = hv * a2;
        #pragma unroll
        for (int s = 16; s > 0; s >>= 1) {
            p1 += __shfl_xor(p1, s, 32);
            p2 += __shfl_xor(p2, s, 32);
        }
        if ((tid & 31) == 0) { atomicAdd(&s1[r], p1); atomicAdd(&s2[r], p2); }
    }
    __syncthreads();

    if (tid < 16) {
        Wh1[rowbase + tid] = s1[tid];
        Wh2[rowbase + tid] = s2[tid];
    }
    if (tid < D) {
        v16h pk;
        #pragma unroll
        for (int rr = 0; rr < 16; ++rr) pk[rr] = (_Float16)hs[rr][tid];
        *(v16h*)(hT + (size_t)tid * N + rowbase) = pk;   // 32B contiguous store
    }
}

// ---------------------------------------------------------------------------
// Kernel 2: global max of Wh2 -> per-row softmax shift m[i]=elu(Wh1[i]+maxWh2)
// (softmax is shift-invariant per row; elu monotonic -> tight upper bound,
//  no adj read needed). Single block.
// ---------------------------------------------------------------------------
__global__ __launch_bounds__(256) void finalize_m_kernel(
    const float* __restrict__ Wh1, const float* __restrict__ Wh2,
    float* __restrict__ m)
{
    __shared__ float red[256];
    const int tid = threadIdx.x;
    float mx = -3.0e38f;
    for (int i = tid; i < N; i += 256) mx = fmaxf(mx, Wh2[i]);
    red[tid] = mx;
    __syncthreads();
    for (int s = 128; s > 0; s >>= 1) {
        if (tid < s) red[tid] = fmaxf(red[tid], red[tid + s]);
        __syncthreads();
    }
    const float gmax = red[0];
    for (int i = tid; i < N; i += 256) {
        float z = Wh1[i] + gmax;
        m[i] = (z > 0.f) ? z : (__expf(z) - 1.f);
    }
}

// ---------------------------------------------------------------------------
// Kernel 3: fused masked-softmax attention + matmul + elu + L2 norm + bias.
// Block: 128 threads (4 waves). Block owns 32 rows (two 16-row A tiles that
// share every B load); wave owns 1/4 of K (split-K via LDS).
// Per 32-wide K step: adj tiles -> P tiles (f32 exp) -> f16 A operands,
// 2 row-sum WMMAs vs all-ones B + 16 tile WMMAs across D=128.
// ---------------------------------------------------------------------------
__global__ __launch_bounds__(128) void gat_attn_kernel(
    const int* __restrict__ adj, const _Float16* __restrict__ hT,
    const float* __restrict__ Wh1, const float* __restrict__ Wh2,
    const float* __restrict__ m, const float* __restrict__ bias,
    float* __restrict__ out)
{
    __shared__ float red4[4][32][D];   // split-K partial accumulators (64 KB)
    __shared__ float psum4[4][32];     // split-K partial row sums
    __shared__ float rsum[32];
    __shared__ float rinv[32];

    const int tid  = threadIdx.x;
    const int wave = tid >> 5;
    const int lane = tid & 31;
    const int r    = lane & 15;        // local row within A/C tile
    const int hi   = lane >> 4;        // lane half selects K sub-range
    const int rowbase = blockIdx.x * 32;
    const int row0 = rowbase + r;
    const int row1 = row0 + 16;

    const float wh1r0 = Wh1[row0], mr0 = m[row0];
    const float wh1r1 = Wh1[row1], mr1 = m[row1];

    const int k_begin = wave * (N / 4);
    const int k_end   = k_begin + (N / 4);
    const int klo     = hi * 8;        // A f16 layout: lanes0-15 K {0..7,16..23}
    const size_t adjrow0 = (size_t)row0 * N;
    const size_t adjrow1 = (size_t)row1 * N;

    v8f acc0[8] = {}, acc1[8] = {};
    v8f asum0 = {}, asum1 = {};

    v16h ones;
    #pragma unroll
    for (int j = 0; j < 16; ++j) ones[j] = (_Float16)1.0f;

    for (int k0 = k_begin; k0 < k_end; k0 += 32) {
        // adjacency tiles for this lane's 16 K positions, both row groups
        int4 a00 = *(const int4*)(adj + adjrow0 + k0 + klo);
        int4 a01 = *(const int4*)(adj + adjrow0 + k0 + klo + 4);
        int4 a02 = *(const int4*)(adj + adjrow0 + k0 + klo + 16);
        int4 a03 = *(const int4*)(adj + adjrow0 + k0 + klo + 20);
        int4 a10 = *(const int4*)(adj + adjrow1 + k0 + klo);
        int4 a11 = *(const int4*)(adj + adjrow1 + k0 + klo + 4);
        int4 a12 = *(const int4*)(adj + adjrow1 + k0 + klo + 16);
        int4 a13 = *(const int4*)(adj + adjrow1 + k0 + klo + 20);
        // prefetch next adj tiles (speculative: OOB at the tail is dropped)
        __builtin_prefetch(adj + adjrow0 + k0 + 32 + klo, 0, 1);
        __builtin_prefetch(adj + adjrow1 + k0 + 32 + klo, 0, 1);

        float4 w0 = *(const float4*)(Wh2 + k0 + klo);
        float4 w1 = *(const float4*)(Wh2 + k0 + klo + 4);
        float4 w2 = *(const float4*)(Wh2 + k0 + klo + 16);
        float4 w3 = *(const float4*)(Wh2 + k0 + klo + 20);

        const float wv[16] = { w0.x, w0.y, w0.z, w0.w, w1.x, w1.y, w1.z, w1.w,
                               w2.x, w2.y, w2.z, w2.w, w3.x, w3.y, w3.z, w3.w };
        const int  av0[16] = { a00.x, a00.y, a00.z, a00.w, a01.x, a01.y, a01.z, a01.w,
                               a02.x, a02.y, a02.z, a02.w, a03.x, a03.y, a03.z, a03.w };
        const int  av1[16] = { a10.x, a10.y, a10.z, a10.w, a11.x, a11.y, a11.z, a11.w,
                               a12.x, a12.y, a12.z, a12.w, a13.x, a13.y, a13.z, a13.w };

        v16h A0, A1;
        #pragma unroll
        for (int j = 0; j < 16; ++j) {
            float z0 = wh1r0 + wv[j];
            float e0 = (z0 > 0.f) ? z0 : (__expf(z0) - 1.f);    // elu
            A0[j] = (_Float16)((av0[j] > 0) ? __expf(e0 - mr0) : 0.f);
            float z1 = wh1r1 + wv[j];
            float e1 = (z1 > 0.f) ? z1 : (__expf(z1) - 1.f);
            A1[j] = (_Float16)((av1[j] > 0) ? __expf(e1 - mr1) : 0.f);
        }

        // row sums via matrix unit: P @ ones -> every column = row sum
        asum0 = __builtin_amdgcn_wmma_f32_16x16x32_f16(
            false, A0, false, ones, (short)0, asum0, false, false);
        asum1 = __builtin_amdgcn_wmma_f32_16x16x32_f16(
            false, A1, false, ones, (short)0, asum1, false, false);

        // 8 B tiles across D=128; each B load feeds two A tiles
        #pragma unroll
        for (int t = 0; t < 8; ++t) {
            const _Float16* bp = hT + (size_t)(t * 16 + r) * N + k0 + hi * 8;
            v16h B;
            #pragma unroll
            for (int j = 0; j < 16; ++j) B[j] = bp[(j < 8) ? j : j + 8];
            acc0[t] = __builtin_amdgcn_wmma_f32_16x16x32_f16(
                false, A0, false, B, (short)0, acc0[t], false, false);
            acc1[t] = __builtin_amdgcn_wmma_f32_16x16x32_f16(
                false, A1, false, B, (short)0, acc1[t], false, false);
        }
    }

    // publish split-K partial row sums (C layout: lane with n==0 holds them)
    if (r == 0) {
        #pragma unroll
        for (int q = 0; q < 8; ++q) {
            psum4[wave][hi * 8 + q]      = asum0[q];
            psum4[wave][16 + hi * 8 + q] = asum1[q];
        }
    }
    // publish split-K partial accumulators (C layout: M = q + 8*hi, n = r)
    #pragma unroll
    for (int t = 0; t < 8; ++t) {
        #pragma unroll
        for (int q = 0; q < 8; ++q) {
            red4[wave][q + hi * 8][t * 16 + r]      = acc0[t][q];
            red4[wave][16 + q + hi * 8][t * 16 + r] = acc1[t][q];
        }
    }
    __syncthreads();

    if (tid < 32)
        rsum[tid] = psum4[0][tid] + psum4[1][tid] + psum4[2][tid] + psum4[3][tid];
    __syncthreads();

    // softmax normalize + elu, per column
    {
        const int c = tid;  // 0..127
        #pragma unroll
        for (int rr = 0; rr < 32; ++rr) {
            float v = red4[0][rr][c] + red4[1][rr][c] + red4[2][rr][c] + red4[3][rr][c];
            v = v / rsum[rr];
            v = (v > 0.f) ? v : (__expf(v) - 1.f);
            red4[0][rr][c] = v;
        }
    }
    __syncthreads();

    if (tid < 32) {
        float ss = 0.f;
        #pragma unroll 8
        for (int c = 0; c < D; ++c) { float v = red4[0][tid][c]; ss += v * v; }
        rinv[tid] = 1.f / fmaxf(sqrtf(ss), 1e-12f);
    }
    __syncthreads();

    {
        const int c = tid;
        const float bc = bias[c];
        #pragma unroll
        for (int rr = 0; rr < 32; ++rr)
            out[(size_t)(rowbase + rr) * D + c] = red4[0][rr][c] * rinv[rr] + bc;
    }
}

// ---------------------------------------------------------------------------
extern "C" void kernel_launch(void* const* d_in, const int* in_sizes, int n_in,
                              void* d_out, int out_size, void* d_ws, size_t ws_size,
                              hipStream_t stream) {
    const float* x      = (const float*)d_in[0];
    const int*   adj    = (const int*)  d_in[1];
    const float* conv_w = (const float*)d_in[2];
    const float* conv_b = (const float*)d_in[3];
    const float* a      = (const float*)d_in[4];
    const float* bias   = (const float*)d_in[5];
    float* out = (float*)d_out;

    char* ws = (char*)d_ws;
    _Float16* hT  = (_Float16*)ws;                       // D*N f16 = 2 MB
    float*    Wh1 = (float*)(ws + (size_t)D * N * 2);    // 32 KB
    float*    Wh2 = Wh1 + N;                             // 32 KB
    float*    mrow= Wh2 + N;                             // 32 KB

    prep_kernel<<<N / 16, 256, 0, stream>>>(x, conv_w, conv_b, a, hT, Wh1, Wh2);
    finalize_m_kernel<<<1, 256, 0, stream>>>(Wh1, Wh2, mrow);
    gat_attn_kernel<<<N / 32, 128, 0, stream>>>(adj, hT, Wh1, Wh2, mrow, bias, out);
}
